// Network_Latent_21251498181075
// MI455X (gfx1250) — matrix operand order repaired and checked by reading, not compiled, not run on hardware
//
#include <hip/hip_runtime.h>

typedef __attribute__((ext_vector_type(16))) _Float16 v16h;
typedef __attribute__((ext_vector_type(8)))  _Float16 v8h;
typedef __attribute__((ext_vector_type(2)))  _Float16 h2;
typedef __attribute__((ext_vector_type(8)))  float    v8f;

#define N_LAYERS 19
#define DIM 6
#define NEG_SLOPE 0.01f

// f32 pair -> packed f16 (v_cvt_pk_f16_f32)
static __device__ __forceinline__ h2 cvt2(float x, float y){
    return (h2){(_Float16)x, (_Float16)y};
}
// f32 pair -> packed f16 with leaky-ReLU in f16 (cvt_pk + pk_mul + pk_max)
static __device__ __forceinline__ h2 lrelu2(float x, float y){
    h2 v = cvt2(x, y);
    return __builtin_elementwise_max(v, v * (h2)(_Float16)NEG_SLOPE);
}
// Build the B operand: data in halves 0..5, everything above zero (zeros must
// stay finite: they multiply structurally-zero A columns and 0*NaN would
// poison the accumulator). Upper half comes from one invariant zero vector.
static __device__ __forceinline__ v16h mkB(h2 p, h2 q, h2 r){
    v8h lo = {p[0], p[1], q[0], q[1], r[0], r[1],
              (_Float16)0.0f, (_Float16)0.0f};
    v8h hi = (v8h)(_Float16)0.0f;
    return __builtin_shufflevector(lo, hi, 0,1,2,3,4,5,6,7,8,9,10,11,12,13,14,15);
}

__global__ __launch_bounds__(256, 5) void mlp_wmma_kernel(
    const float* __restrict__ X,      // [nrows,6]
    const float* __restrict__ Ws,     // [19,6,6]
    const float* __restrict__ bs,     // [19,6]
    const float* __restrict__ bounds, // [6,2]
    float* __restrict__ out,          // [nrows,6]
    int nrows)
{
    // Per-lane A-operand image (f16 pairs) per layer: [layer][lane][8 dwords]
    __shared__ alignas(32) unsigned int sA[N_LAYERS*32*8];
    // Lane-uniform bias image for the C operand: [layer][8 floats]
    __shared__ alignas(32) float sC[N_LAYERS*8];
    __shared__ float sIs[DIM], sNl[DIM];

    const int tid = threadIdx.x;

    // ---- Build block-diagonal A images in LDS ----
    // A (16x32 f16): lanes 0-15 -> M=lane,    K=0..15 in V0..V7
    //                lanes16-31 -> M=lane-16, K=16..31 in V0..V7
    // Block 1: A[m,k] = W[m,k] (m<6,k<6); Block 2: A[m+8,k+16] = W[m,k]
    for (int idx = tid; idx < N_LAYERS*32*8; idx += 256) {
        int L   = idx >> 8;
        int rem = idx & 255;
        int lane = rem >> 3;
        int j    = rem & 7;
        int grp  = lane >> 4;
        int m    = lane & 15;
        int kb   = grp*16 + 2*j;
        unsigned int w = 0;
        #pragma unroll
        for (int h = 0; h < 2; ++h) {
            int k = kb + h;
            float v = 0.0f;
            if (m < 6 && k < 6)                             v = Ws[L*36 + m*6 + k];
            else if (m >= 8 && m < 14 && k >= 16 && k < 22) v = Ws[L*36 + (m-8)*6 + (k-16)];
            unsigned short hb = __builtin_bit_cast(unsigned short, (_Float16)v);
            w |= ((unsigned int)hb) << (16*h);
        }
        sA[idx] = w;
    }
    for (int idx = tid; idx < N_LAYERS*8; idx += 256) {
        int L = idx >> 3; int v = idx & 7;
        sC[idx] = (v < 6) ? bs[L*6 + v] : 0.0f;
    }
    if (tid < DIM) {
        float lo = bounds[2*tid], hi = bounds[2*tid+1];
        float is = 1.0f/(hi - lo);
        sIs[tid] = is;
        sNl[tid] = -lo*is;
    }
    __syncthreads();

    const int lane = tid & 31;
    const float2* X2 = (const float2*)X;
    float2* O2 = (float2*)out;
    const int stride = gridDim.x * blockDim.x;

    const v16h* pA = (const v16h*)sA;  // 32B-aligned, 32B stride
    const v8f*  pC = (const v8f*)sC;   // 32B-aligned, 32B stride

    float is0=sIs[0],is1=sIs[1],is2=sIs[2],is3=sIs[3],is4=sIs[4],is5=sIs[5];
    float nl0=sNl[0],nl1=sNl[1],nl2=sNl[2],nl3=sNl[3],nl4=sNl[4],nl5=sNl[5];

    struct Chain {
        v16h bv; v8f d;
        float r0,r1,r2,r3,r4,r5;
        int row;
    };

    auto initChain = [&](Chain& ch, int row){
        ch.row = row;
        float2 a0 = X2[row*3+0], a1 = X2[row*3+1], a2 = X2[row*3+2];
        float w0=fmaf(a0.x,is0,nl0), w1=fmaf(a0.y,is1,nl1);
        float w2=fmaf(a1.x,is2,nl2), w3=fmaf(a1.y,is3,nl3);
        float w4=fmaf(a2.x,is4,nl4), w5=fmaf(a2.y,is5,nl5);
        ch.bv = mkB(cvt2(w0,w1), cvt2(w2,w3), cvt2(w4,w5));
        ch.r0=w0; ch.r1=w1; ch.r2=w2; ch.r3=w3; ch.r4=w4; ch.r5=w5;
    };

    // variant: 0 = leaky-relu, 1 = residual add, 2 = final store
    auto post = [&](Chain& ch, int variant){
        if (variant == 2) {
            O2[ch.row*3+0] = make_float2(ch.d[0], ch.d[1]);
            O2[ch.row*3+1] = make_float2(ch.d[2], ch.d[3]);
            O2[ch.row*3+2] = make_float2(ch.d[4], ch.d[5]);
        } else if (variant == 1) {
            ch.d[0]+=ch.r0; ch.d[1]+=ch.r1; ch.d[2]+=ch.r2;
            ch.d[3]+=ch.r3; ch.d[4]+=ch.r4; ch.d[5]+=ch.r5;
            ch.r0=ch.d[0]; ch.r1=ch.d[1]; ch.r2=ch.d[2];
            ch.r3=ch.d[3]; ch.r4=ch.d[4]; ch.r5=ch.d[5];
            ch.bv = mkB(cvt2(ch.d[0],ch.d[1]), cvt2(ch.d[2],ch.d[3]),
                        cvt2(ch.d[4],ch.d[5]));
        } else {
            ch.bv = mkB(lrelu2(ch.d[0],ch.d[1]), lrelu2(ch.d[2],ch.d[3]),
                        lrelu2(ch.d[4],ch.d[5]));
        }
    };

    // Three interleaved row-chains sharing each layer's A/C operands.
    auto body3 = [&](int rA, int rB, int rC, unsigned lidx, unsigned cidx){
        Chain A, B, C;
        initChain(A, rA); initChain(B, rB); initChain(C, rC);
        __builtin_prefetch(&X2[(long long)rA*3 + (long long)stride*9], 0, 0);
        __builtin_prefetch(&X2[(long long)rB*3 + (long long)stride*9], 0, 0);
        __builtin_prefetch(&X2[(long long)rC*3 + (long long)stride*9], 0, 0);

        // One pipeline step: finish layer L-1 (post, variant pv) for each
        // chain right before issuing that chain's layer-L WMMA. Every
        // WMMA->VALU RAW gets ~2 WMMAs + one post in its shadow.
        auto step = [&](unsigned aIdx, unsigned cIdx, int pv){
            v16h a = pA[aIdx];   // 2x ds_load_b128, per-layer imm offset
            v8f  c = pC[cIdx];   // 2x ds_load_b128
            if (pv >= 0) post(A, pv);
            A.d = __builtin_amdgcn_wmma_f32_16x16x32_f16(false, a, false, A.bv,
                                                         (short)0, c, false, false);
            if (pv >= 0) post(B, pv);
            B.d = __builtin_amdgcn_wmma_f32_16x16x32_f16(false, a, false, B.bv,
                                                         (short)0, c, false, false);
            if (pv >= 0) post(C, pv);
            C.d = __builtin_amdgcn_wmma_f32_16x16x32_f16(false, a, false, C.bv,
                                                         (short)0, c, false, false);
        };

        step(lidx, cidx, -1);                      // layer 0
        unsigned aI = lidx, cI = cidx;
        // Real loop (4 layers per iteration) to cap the scheduling window and
        // keep register pressure bounded. Layers 4b+1 .. 4b+4; previous-layer
        // post variants are compile-time: lrelu,lrelu,lrelu,residual(4b+3).
        #pragma unroll 1
        for (int b = 0; b < 4; ++b) {
            step(aI + 1*32, cI + 1, 0);
            step(aI + 2*32, cI + 2, 0);
            step(aI + 3*32, cI + 3, 0);
            step(aI + 4*32, cI + 4, 1);
            aI += 4*32; cI += 4;
        }
        step(aI + 1*32, cI + 1, 0);                // layer 17 (post 16: lrelu)
        step(aI + 2*32, cI + 2, 0);                // layer 18 (post 17: lrelu)
        post(A, 2); post(B, 2); post(C, 2);        // stores
    };

    auto body1 = [&](int r, unsigned lidx, unsigned cidx){
        Chain A;
        initChain(A, r);
        auto step1 = [&](unsigned aIdx, unsigned cIdx, int pv){
            v16h a = pA[aIdx];
            v8f  c = pC[cIdx];
            if (pv >= 0) post(A, pv);
            A.d = __builtin_amdgcn_wmma_f32_16x16x32_f16(false, a, false, A.bv,
                                                         (short)0, c, false, false);
        };
        step1(lidx, cidx, -1);
        unsigned aI = lidx, cI = cidx;
        #pragma unroll 1
        for (int b = 0; b < 4; ++b) {
            step1(aI + 1*32, cI + 1, 0);
            step1(aI + 2*32, cI + 2, 0);
            step1(aI + 3*32, cI + 3, 0);
            step1(aI + 4*32, cI + 4, 1);
            aI += 4*32; cI += 4;
        }
        step1(aI + 1*32, cI + 1, 0);
        step1(aI + 2*32, cI + 2, 0);
        post(A, 2);
    };

    int row = blockIdx.x*blockDim.x + tid;
    // Main loop: triples (row, row+stride, row+2*stride).
    for (; row + 2*stride < nrows; row += 3*stride) {
        // Launder the LDS indices (32B units -> alignment stays provable) so
        // weight loads cannot be hoisted across row-chunk iterations.
        unsigned lidx = (unsigned)lane, cidx = 0;
        asm volatile("" : "+v"(lidx), "+v"(cidx));
        body3(row, row + stride, row + 2*stride, lidx, cidx);
    }
    // Tail: remaining strides, one chain at a time (wave-uniform EXEC).
    for (; row < nrows; row += stride) {
        unsigned lidx = (unsigned)lane, cidx = 0;
        asm volatile("" : "+v"(lidx), "+v"(cidx));
        body1(row, lidx, cidx);
    }
}

extern "C" void kernel_launch(void* const* d_in, const int* in_sizes, int n_in,
                              void* d_out, int out_size, void* d_ws, size_t ws_size,
                              hipStream_t stream) {
    const float* X      = (const float*)d_in[0];
    const float* Ws     = (const float*)d_in[1];
    const float* bs     = (const float*)d_in[2];
    const float* bounds = (const float*)d_in[3];
    float* out = (float*)d_out;

    int nrows = in_sizes[0] / DIM;   // 4,194,304
    int block = 256;                 // 8 wave32 per workgroup
    int grid  = 1024;                // triples: 96 rows/wave/iter
    if (grid * block > nrows) grid = (nrows + block - 1) / block;

    mlp_wmma_kernel<<<grid, block, 0, stream>>>(X, Ws, bs, bounds, out, nrows);
}
